// UniCoreLinearGroupFP8_8504035246562
// MI455X (gfx1250) — 8-bit matrix operand order repaired and checked by reading, compiled, not run on hardware
//
#include <hip/hip_runtime.h>
#include <hip/hip_bf16.h>

typedef __attribute__((ext_vector_type(16))) int   v16i;
typedef __attribute__((ext_vector_type(8)))  float v8f;

typedef int vs4i __attribute__((vector_size(16)));
typedef __attribute__((address_space(1))) vs4i as1_v4i;
typedef __attribute__((address_space(3))) vs4i as3_v4i;

#define GS   128
#define GN   32
#define BM   128
#define BN   64
#define TILE_A_BYTES (BM * GS)
#define TILE_B_BYTES (BN * GS)

#if __has_builtin(__builtin_amdgcn_global_load_async_to_lds_b128)
#define HAS_ASYNC 1
#else
#define HAS_ASYNC 0
#endif

__device__ __forceinline__ unsigned char bf16_to_e4m3(unsigned short bits) {
  unsigned s = (bits >> 8) & 0x80u;
  unsigned a = bits & 0x7FFFu;
  if (a == 0u) return (unsigned char)s;
  union { unsigned u; float f; } cv;
  cv.u = a << 16;
  float absw = cv.f;
  if (absw < 0.015625f) {
    int m = (int)__builtin_rintf(absw * 512.0f);
    if (m > 7) m = 7;
    return (unsigned char)(s | (unsigned)m);
  }
  int e = (int)((bits >> 7) & 0xFFu) - 127;
  e = e < -6 ? -6 : (e > 8 ? 8 : e);
  unsigned e4 = ((unsigned)(e + 7) & 15u) << 3;
  unsigned m3 = (bits >> 4) & 7u;
  return (unsigned char)(s | e4 | m3);
}

__global__ void __launch_bounds__(256) quant_bf16_to_fp8(
    const unsigned short* __restrict__ in, unsigned char* __restrict__ out,
    long long n8) {
  long long i = (long long)blockIdx.x * 256 + threadIdx.x;
  if (i >= n8) return;
  uint4 raw = *(const uint4*)(in + i * 8);
  unsigned short h[8];
  h[0] = (unsigned short)raw.x; h[1] = (unsigned short)(raw.x >> 16);
  h[2] = (unsigned short)raw.y; h[3] = (unsigned short)(raw.y >> 16);
  h[4] = (unsigned short)raw.z; h[5] = (unsigned short)(raw.z >> 16);
  h[6] = (unsigned short)raw.w; h[7] = (unsigned short)(raw.w >> 16);
  unsigned lo = 0, hi = 0;
#pragma unroll
  for (int j = 0; j < 4; ++j) lo |= (unsigned)bf16_to_e4m3(h[j]) << (8 * j);
#pragma unroll
  for (int j = 0; j < 4; ++j) hi |= (unsigned)bf16_to_e4m3(h[4 + j]) << (8 * j);
  uint2 r; r.x = lo; r.y = hi;
  *(uint2*)(out + i * 8) = r;
}

__device__ __forceinline__ void cp16(const unsigned char* g, unsigned char* l) {
#if HAS_ASYNC
  __builtin_amdgcn_global_load_async_to_lds_b128(
      (as1_v4i*)g, (as3_v4i*)l, 0, 0);
#else
  *(uint4*)l = *(const uint4*)g;
#endif
}

__device__ __forceinline__ void wait_tiles() {
#if HAS_ASYNC
#if __has_builtin(__builtin_amdgcn_s_wait_asynccnt)
  __builtin_amdgcn_s_wait_asynccnt(0);
#else
  asm volatile("s_wait_asynccnt 0" ::: "memory");
#endif
#endif
}

__device__ __forceinline__ void prefetch_tiles(
    const unsigned char* __restrict__ Aq, const unsigned char* __restrict__ Wq,
    unsigned char* As, unsigned char* Bs, int bm, int bn, int kg, int K, int tid) {
  const unsigned char* Ag = Aq + (long long)bm * K + (long long)kg * GS;
  const unsigned char* Bg = Wq + (long long)bn * K + (long long)kg * GS;
#pragma unroll
  for (int i = 0; i < 4; ++i) {
    int idx = tid + i * 256;
    int r = idx >> 3, c = idx & 7;
    cp16(Ag + (long long)r * K + c * 16, As + r * GS + ((c ^ (r & 7)) << 4));
  }
#pragma unroll
  for (int i = 0; i < 2; ++i) {
    int idx = tid + i * 256;
    int r = idx >> 3, c = idx & 7;
    cp16(Bg + (long long)r * K + c * 16, Bs + r * GS + ((c ^ (r & 7)) << 4));
  }
}

__device__ __forceinline__ float bf16u_to_f(unsigned short u) {
  union { unsigned v; float f; } c;
  c.v = ((unsigned)u) << 16;
  return c.f;
}

__global__ void __launch_bounds__(256) gemm_fp8_group(
    const unsigned char* __restrict__ Aq, const unsigned char* __restrict__ Wq,
    const unsigned short* __restrict__ xscale,
    const unsigned short* __restrict__ wscale,
    __hip_bfloat16* __restrict__ out, int M, int N, int K) {
  __shared__ __align__(16) unsigned char smem[2 * TILE_A_BYTES + 2 * TILE_B_BYTES +
                                              BM * GN * 2 + BN * GN * 2];
  unsigned short* sxs = (unsigned short*)(smem + 2 * TILE_A_BYTES + 2 * TILE_B_BYTES);
  unsigned short* sws = sxs + BM * GN;

  const int tid = threadIdx.x;
  const int bn = blockIdx.x * BN;
  const int bm = blockIdx.y * BM;
  const int G = K / GS;

#pragma unroll
  for (int i = 0; i < 2; ++i) {
    int idx = tid + i * 256;
    int r = idx >> 2, c = idx & 3;
    *(uint4*)((unsigned char*)sxs + r * (GN * 2) + c * 16) =
        *(const uint4*)((const unsigned char*)xscale + (long long)(bm + r) * (G * 2) + c * 16);
  }
  {
    int r = tid >> 2, c = tid & 3;
    *(uint4*)((unsigned char*)sws + r * (GN * 2) + c * 16) =
        *(const uint4*)((const unsigned char*)wscale + (long long)(bn + r) * (G * 2) + c * 16);
  }

  const int wv = tid >> 5;
  const int lane = tid & 31;
  const int L = lane & 15;
  const int hi = lane >> 4;
  const int wm = (wv >> 1) * 32;
  const int wn = (wv & 1) * 32;

  float acc[2][2][8];
#pragma unroll
  for (int mt = 0; mt < 2; ++mt)
#pragma unroll
    for (int nt = 0; nt < 2; ++nt)
#pragma unroll
      for (int v = 0; v < 8; ++v) acc[mt][nt][v] = 0.0f;

  prefetch_tiles(Aq, Wq, smem, smem + 2 * TILE_A_BYTES, bm, bn, 0, K, tid);

  for (int kg = 0; kg < G; ++kg) {
    const int cur = kg & 1;
    const int nxt = cur ^ 1;
    wait_tiles();
    __syncthreads();
    if (kg + 1 < G)
      prefetch_tiles(Aq, Wq,
                     smem + nxt * TILE_A_BYTES,
                     smem + 2 * TILE_A_BYTES + nxt * TILE_B_BYTES,
                     bm, bn, kg + 1, K, tid);

    const unsigned char* Ac = smem + cur * TILE_A_BYTES;
    const unsigned char* Bc = smem + 2 * TILE_A_BYTES + cur * TILE_B_BYTES;

    v16i af[2];
#pragma unroll
    for (int mt = 0; mt < 2; ++mt) {
      int r0 = wm + mt * 16 + L;
#pragma unroll
      for (int c = 0; c < 4; ++c) {
        int cc = 2 * c + hi;
        uint4 d = *(const uint4*)(Ac + r0 * GS + ((cc ^ (r0 & 7)) << 4));
        af[mt][4 * c]     = (int)d.x;
        af[mt][4 * c + 1] = (int)d.y;
        af[mt][4 * c + 2] = (int)d.z;
        af[mt][4 * c + 3] = (int)d.w;
      }
    }
    v16i bfr[2];
#pragma unroll
    for (int nt = 0; nt < 2; ++nt) {
      int r0 = wn + nt * 16 + L;
#pragma unroll
      for (int c = 0; c < 4; ++c) {
        int cc = 2 * c + hi;
        uint4 d = *(const uint4*)(Bc + r0 * GS + ((cc ^ (r0 & 7)) << 4));
        bfr[nt][4 * c]     = (int)d.x;
        bfr[nt][4 * c + 1] = (int)d.y;
        bfr[nt][4 * c + 2] = (int)d.z;
        bfr[nt][4 * c + 3] = (int)d.w;
      }
    }

    float sxf[2][8], swf[2];
#pragma unroll
    for (int mt = 0; mt < 2; ++mt)
#pragma unroll
      for (int v = 0; v < 8; ++v) {
        int ml = wm + mt * 16 + hi * 8 + v;
        sxf[mt][v] = bf16u_to_f(sxs[ml * GN + kg]);
      }
#pragma unroll
    for (int nt = 0; nt < 2; ++nt) {
      int nl = wn + nt * 16 + L;
      swf[nt] = bf16u_to_f(sws[nl * GN + kg]);
    }

    v8f z = {0.f, 0.f, 0.f, 0.f, 0.f, 0.f, 0.f, 0.f};
#pragma unroll
    for (int mt = 0; mt < 2; ++mt)
#pragma unroll
      for (int nt = 0; nt < 2; ++nt) {
        v8f part = __builtin_amdgcn_wmma_f32_16x16x128_fp8_fp8(
            af[mt], bfr[nt], (short)0, z, false, false);
#pragma unroll
        for (int v = 0; v < 8; ++v)
          acc[mt][nt][v] = fmaf(part[v], sxf[mt][v] * swf[nt], acc[mt][nt][v]);
      }
  }

#pragma unroll
  for (int mt = 0; mt < 2; ++mt)
#pragma unroll
    for (int nt = 0; nt < 2; ++nt) {
      int n = bn + wn + nt * 16 + L;
#pragma unroll
      for (int v = 0; v < 8; ++v) {
        int m = bm + wm + mt * 16 + hi * 8 + v;
        out[(long long)m * N + n] = __float2bfloat16(acc[mt][nt][v]);
      }
    }
}

extern "C" void kernel_launch(void* const* d_in, const int* in_sizes, int n_in,
                              void* d_out, int out_size, void* d_ws, size_t ws_size,
                              hipStream_t stream) {
  (void)n_in; (void)out_size; (void)ws_size;
  const int K = 4096;
  const long long xTot = in_sizes[0];
  const long long wTot = in_sizes[2];
  const int M = (int)(xTot / K);
  const int N = (int)(wTot / K);

  unsigned char* Aq = (unsigned char*)d_ws;
  unsigned char* Wq = Aq + xTot;

  const long long nx8 = xTot >> 3;
  const long long nw8 = wTot >> 3;
  quant_bf16_to_fp8<<<(unsigned)((nx8 + 255) / 256), 256, 0, stream>>>(
      (const unsigned short*)d_in[0], Aq, nx8);
  quant_bf16_to_fp8<<<(unsigned)((nw8 + 255) / 256), 256, 0, stream>>>(
      (const unsigned short*)d_in[2], Wq, nw8);

  dim3 grid(N / BN, M / BM);
  gemm_fp8_group<<<grid, 256, 0, stream>>>(
      Aq, Wq, (const unsigned short*)d_in[1], (const unsigned short*)d_in[3],
      (__hip_bfloat16*)d_out, M, N, K);
}
